// MAGNN_metapath_specific_73598559584366
// MI455X (gfx1250) — matrix operand order repaired and checked by reading, not compiled, not run on hardware
//
#include <hip/hip_runtime.h>
#include <hip/hip_bf16.h>

typedef __attribute__((ext_vector_type(16))) _Float16 v16h;
typedef __attribute__((ext_vector_type(8)))  float    v8f;

#define SEQL 5
#define DM   64
#define NH   8
#define DH   8
#define EB   16          // edges per workgroup
#define ROWS (EB * SEQL) // 80 token rows per block

// ---------------- wave / math helpers ----------------

__device__ __forceinline__ float wave_sum32(float v) {
#pragma unroll
  for (int o = 16; o > 0; o >>= 1) v += __shfl_xor(v, o, 32);
  return v;
}

__device__ __forceinline__ float gelu_erf(float x) {
  return 0.5f * x * (1.0f + erff(x * 0.70710678118654752f));
}

__device__ __forceinline__ float leaky01(float x) { return x > 0.f ? x : 0.01f * x; }

// order-preserving float<->uint for atomicMax over signed floats
__device__ __forceinline__ unsigned fenc(float f) {
  unsigned u = __float_as_uint(f);
  return (u & 0x80000000u) ? ~u : (u | 0x80000000u);
}
__device__ __forceinline__ float fdec(unsigned u) {
  u = (u & 0x80000000u) ? (u & 0x7fffffffu) : ~u;
  return __uint_as_float(u);
}

// ---------------- WMMA fragment loaders (CDNA5 16x16x32 f16) ----------------
// A (16x32, f16): lane l: m=l&15, h=l>>4; VGPR v holds K pair at
//   kk + (v>>2)*16 + h*8 + (v&3)*2   (ISA 7.12.2, 16-bit A 16x32)
// B (32x16, f16): lane l: n=l&15, h=l>>4; VGPR v holds K pair at
//   kk + (v>>2)*16 + h*8 + (v&3)*2   (same K order as A)
// C/D (16x16 f32): lane l: n=l&15; VGPR v: m=(l>>4)*8+v

union HFrag { v16h v; unsigned u[8]; };

__device__ __forceinline__ v16h load_a_frag(const _Float16* base, int ld, int mBase, int kk) {
  const int l = threadIdx.x & 31;
  const int m = l & 15, h = l >> 4;
  HFrag f;
#pragma unroll
  for (int r = 0; r < 8; ++r) {
    int k = kk + (r >> 2) * 16 + h * 8 + (r & 3) * 2;
    f.u[r] = *(const unsigned*)(base + (mBase + m) * ld + k);
  }
  return f.v;
}

// W is row-major [Nout][K] fp16; B[k][n] = W[nBase+n][k]
__device__ __forceinline__ v16h load_b_frag(const _Float16* W, int ldk, int nBase, int kk) {
  const int l = threadIdx.x & 31;
  const int n = l & 15, h = l >> 4;
  HFrag f;
#pragma unroll
  for (int r = 0; r < 8; ++r) {
    int k = kk + (r >> 2) * 16 + h * 8 + (r & 3) * 2;
    f.u[r] = *(const unsigned*)(W + (nBase + n) * ldk + k);
  }
  return f.v;
}

__device__ __forceinline__ v8f wmma16(v16h a, v16h b, v8f c) {
  return __builtin_amdgcn_wmma_f32_16x16x32_f16(false, a, false, b, (short)0, c, false, false);
}

// ---------------- utility kernels ----------------

__global__ __launch_bounds__(256) void cvt_kernel(const float* __restrict__ s,
                                                  _Float16* __restrict__ d, int n) {
  int i = blockIdx.x * 256 + threadIdx.x;
  if (i < n) d[i] = (_Float16)s[i];
}

__global__ __launch_bounds__(256) void zero_kernel(unsigned* __restrict__ p, long long n) {
  long long i = (long long)blockIdx.x * 256 + threadIdx.x;
  if (i < n) p[i] = 0u;
}

// tokens[e*5+s][c] = feats[idx] + pos[s]  (zero for pads: idx==N)
__global__ __launch_bounds__(256) void gather_tokens_kernel(
    const float* __restrict__ feats, const float* __restrict__ pos,
    const int* __restrict__ idx, float* __restrict__ tokens, int E, int N) {
  int i = blockIdx.x * 256 + threadIdx.x;
  if (i >= E * SEQL * DM) return;
  int t = i >> 6, c = i & 63;
  int s = t % SEQL;
  int id = idx[t];
  float v = 0.f;
  if (id >= 0 && id < N) v = feats[(size_t)id * DM + c] + pos[s * DM + c];
  tokens[i] = v;
}

// ---------------- fused self-attention block (16 edges / block) ----------------

__global__ __launch_bounds__(256) void self_attn_kernel(
    float* __restrict__ tokens, const int* __restrict__ idx,
    const _Float16* __restrict__ Wqkv, const float* __restrict__ bqkv,
    const _Float16* __restrict__ Wout, const float* __restrict__ bout,
    const float* __restrict__ ln_g, const float* __restrict__ ln_b, int E, int N) {
  __shared__ _Float16 sX[ROWS * DM];     // LN'ed tokens (A of QKV GEMM)
  __shared__ _Float16 sQKV[ROWS * 192];  // q|k|v
  __shared__ _Float16 sAO[ROWS * DM];    // attention output (A of out-proj)
  const int e0 = blockIdx.x * EB;
  const int wave = threadIdx.x >> 5, lane = threadIdx.x & 31;
  __builtin_prefetch(tokens + (size_t)e0 * SEQL * DM, 0, 0);

  // LayerNorm of 80 token rows (one wave per row, lane holds 2 elems)
  const float g0 = ln_g[2 * lane], g1 = ln_g[2 * lane + 1];
  const float bb0 = ln_b[2 * lane], bb1 = ln_b[2 * lane + 1];
  for (int r = wave; r < ROWS; r += 8) {
    const float* tp = tokens + (size_t)(e0 * SEQL + r) * DM;
    float x0 = tp[2 * lane], x1 = tp[2 * lane + 1];
    float m = wave_sum32(x0 + x1) * (1.f / 64.f);
    float d0 = x0 - m, d1 = x1 - m;
    float var = wave_sum32(d0 * d0 + d1 * d1) * (1.f / 64.f);
    float rs = rsqrtf(var + 1e-5f);
    sX[r * DM + 2 * lane]     = (_Float16)(d0 * rs * g0 + bb0);
    sX[r * DM + 2 * lane + 1] = (_Float16)(d1 * rs * g1 + bb1);
  }
  __syncthreads();

  // QKV GEMM: [80,64] x W^T[64,192] -> 5x12 tiles over 8 waves
  for (int t = wave; t < 60; t += 8) {
    int mt = t / 12, nt = t % 12;
    v8f acc = {};
#pragma unroll
    for (int kk = 0; kk < 64; kk += 32) {
      v16h a = load_a_frag(sX, DM, mt * 16, kk);
      v16h b = load_b_frag(Wqkv, DM, nt * 16, kk);
      acc = wmma16(a, b, acc);
    }
    int n = nt * 16 + (lane & 15);
    float bias = bqkv[n];
    int hh = lane >> 4;
#pragma unroll
    for (int r = 0; r < 8; ++r) {
      int m = mt * 16 + hh * 8 + r;
      sQKV[m * 192 + n] = (_Float16)(acc[r] + bias);
    }
  }
  __syncthreads();

  // per-(edge,head) 5x5 attention in VALU
  if (threadIdx.x < EB * NH) {
    int el = threadIdx.x >> 3, hh = threadIdx.x & 7;
    int e = e0 + el;
    bool valid[SEQL];
#pragma unroll
    for (int s = 0; s < SEQL; ++s) valid[s] = (idx[e * SEQL + s] < N);
    const float scale = 0.35355339059327373f;  // dh^-0.5, dh=8
#pragma unroll
    for (int sq = 0; sq < SEQL; ++sq) {
      float qv[DH];
#pragma unroll
      for (int j = 0; j < DH; ++j)
        qv[j] = (float)sQKV[(el * SEQL + sq) * 192 + hh * DH + j] * scale;
      float sc[SEQL];
      float mx = -1e30f;
#pragma unroll
      for (int sk = 0; sk < SEQL; ++sk) {
        float dd = 0.f;
#pragma unroll
        for (int j = 0; j < DH; ++j)
          dd += qv[j] * (float)sQKV[(el * SEQL + sk) * 192 + 64 + hh * DH + j];
        sc[sk] = valid[sk] ? dd : -1e9f;
        mx = fmaxf(mx, sc[sk]);
      }
      float den = 0.f;
#pragma unroll
      for (int sk = 0; sk < SEQL; ++sk) { sc[sk] = __expf(sc[sk] - mx); den += sc[sk]; }
      float inv = 1.f / den;
#pragma unroll
      for (int j = 0; j < DH; ++j) {
        float o = 0.f;
#pragma unroll
        for (int sk = 0; sk < SEQL; ++sk)
          o += sc[sk] * (float)sQKV[(el * SEQL + sk) * 192 + 128 + hh * DH + j];
        sAO[(el * SEQL + sq) * DM + hh * DH + j] = (_Float16)(o * inv);
      }
    }
  }
  __syncthreads();

  // out-proj GEMM [80,64]x[64,64] + residual into tokens (in place)
  for (int t = wave; t < 20; t += 8) {
    int mt = t / 4, nt = t % 4;
    v8f acc = {};
#pragma unroll
    for (int kk = 0; kk < 64; kk += 32) {
      v16h a = load_a_frag(sAO, DM, mt * 16, kk);
      v16h b = load_b_frag(Wout, DM, nt * 16, kk);
      acc = wmma16(a, b, acc);
    }
    int n = nt * 16 + (lane & 15);
    float bias = bout[n];
    int hh = lane >> 4;
#pragma unroll
    for (int r = 0; r < 8; ++r) {
      int m = mt * 16 + hh * 8 + r;
      size_t gi = (size_t)(e0 * SEQL + m) * DM + n;
      tokens[gi] = tokens[gi] + acc[r] + bias;
    }
  }
}

// ---------------- masked mean-pool + FF (sa) ----------------

__global__ __launch_bounds__(256) void pool_ff_kernel(
    const float* __restrict__ tokens, const int* __restrict__ idx, float* __restrict__ inst,
    const float* __restrict__ ln_g, const float* __restrict__ ln_b,
    const _Float16* __restrict__ W1, const float* __restrict__ b1,
    const _Float16* __restrict__ W2, const float* __restrict__ b2, int E, int N) {
  __shared__ float sI[EB * DM];
  __shared__ _Float16 sA[EB * DM];
  __shared__ _Float16 sH[EB * 256];
  const int e0 = blockIdx.x * EB;
  const int wave = threadIdx.x >> 5, lane = threadIdx.x & 31;

  for (int i = threadIdx.x; i < EB * DM; i += 256) {
    int el = i >> 6, c = i & 63;
    int e = e0 + el;
    float s = 0.f, cnt = 0.f;
#pragma unroll
    for (int t = 0; t < SEQL; ++t)
      if (idx[e * SEQL + t] < N) { s += tokens[(size_t)(e * SEQL + t) * DM + c]; cnt += 1.f; }
    sI[i] = s / fmaxf(cnt, 1.f);
  }
  __syncthreads();
  {
    const float g0 = ln_g[2 * lane], g1 = ln_g[2 * lane + 1];
    const float bb0 = ln_b[2 * lane], bb1 = ln_b[2 * lane + 1];
    for (int r = wave; r < EB; r += 8) {
      float x0 = sI[r * DM + 2 * lane], x1 = sI[r * DM + 2 * lane + 1];
      float m = wave_sum32(x0 + x1) * (1.f / 64.f);
      float d0 = x0 - m, d1 = x1 - m;
      float var = wave_sum32(d0 * d0 + d1 * d1) * (1.f / 64.f);
      float rs = rsqrtf(var + 1e-5f);
      sA[r * DM + 2 * lane]     = (_Float16)(d0 * rs * g0 + bb0);
      sA[r * DM + 2 * lane + 1] = (_Float16)(d1 * rs * g1 + bb1);
    }
  }
  __syncthreads();
  // FF1: [16,64] x [64,256], GELU epilogue
  for (int nt = wave; nt < 16; nt += 8) {
    v8f acc = {};
#pragma unroll
    for (int kk = 0; kk < 64; kk += 32) {
      v16h a = load_a_frag(sA, DM, 0, kk);
      v16h b = load_b_frag(W1, DM, nt * 16, kk);
      acc = wmma16(a, b, acc);
    }
    int n = nt * 16 + (lane & 15);
    float bias = b1[n];
    int hh = lane >> 4;
#pragma unroll
    for (int r = 0; r < 8; ++r) sH[(hh * 8 + r) * 256 + n] = (_Float16)gelu_erf(acc[r] + bias);
  }
  __syncthreads();
  // FF2: [16,256] x [256,64] + residual -> inst
  if (wave < 4) {
    int nt = wave;
    v8f acc = {};
#pragma unroll
    for (int kk = 0; kk < 256; kk += 32) {
      v16h a = load_a_frag(sH, 256, 0, kk);
      v16h b = load_b_frag(W2, 256, nt * 16, kk);
      acc = wmma16(a, b, acc);
    }
    int n = nt * 16 + (lane & 15);
    float bias = b2[n];
    int hh = lane >> 4;
#pragma unroll
    for (int r = 0; r < 8; ++r) {
      int m = hh * 8 + r;
      inst[(size_t)(e0 + m) * DM + n] = sI[m * DM + n] + acc[r] + bias;
    }
  }
}

// ---------------- cross-attention + FF (ca) ----------------

__global__ __launch_bounds__(256) void cross_ff_kernel(
    const float* __restrict__ tokens, const float* __restrict__ feats,
    const int* __restrict__ idx, float* __restrict__ inst,
    const float* __restrict__ ln1_g, const float* __restrict__ ln1_b,
    const _Float16* __restrict__ Wci, const float* __restrict__ bci,
    const _Float16* __restrict__ Wco, const float* __restrict__ bco,
    const float* __restrict__ ln2_g, const float* __restrict__ ln2_b,
    const _Float16* __restrict__ W1, const float* __restrict__ b1,
    const _Float16* __restrict__ W2, const float* __restrict__ b2, int E, int N) {
  __shared__ _Float16 sQ[EB * DM];
  __shared__ _Float16 sT[ROWS * DM];
  __shared__ float sQp[EB * DM];
  __shared__ _Float16 sKV[ROWS * 128];
  __shared__ _Float16 sCO[EB * DM];
  __shared__ float sI[EB * DM];
  __shared__ _Float16 sA[EB * DM];
  __shared__ _Float16 sH[EB * 256];
  const int e0 = blockIdx.x * EB;
  const int wave = threadIdx.x >> 5, lane = threadIdx.x & 31;

  // q rows: LN(target node feature)
  {
    const float g0 = ln1_g[2 * lane], g1 = ln1_g[2 * lane + 1];
    const float bb0 = ln1_b[2 * lane], bb1 = ln1_b[2 * lane + 1];
    for (int r = wave; r < EB; r += 8) {
      int e = e0 + r;
      int len = 0;
#pragma unroll
      for (int s = 0; s < SEQL; ++s) len += (idx[e * SEQL + s] < N) ? 1 : 0;
      int lp = len > 0 ? len - 1 : 0;
      int id = idx[e * SEQL + lp];
      float x0 = 0.f, x1 = 0.f;
      if (id >= 0 && id < N) {
        x0 = feats[(size_t)id * DM + 2 * lane];
        x1 = feats[(size_t)id * DM + 2 * lane + 1];
      }
      float m = wave_sum32(x0 + x1) * (1.f / 64.f);
      float d0 = x0 - m, d1 = x1 - m;
      float var = wave_sum32(d0 * d0 + d1 * d1) * (1.f / 64.f);
      float rs = rsqrtf(var + 1e-5f);
      sQ[r * DM + 2 * lane]     = (_Float16)(d0 * rs * g0 + bb0);
      sQ[r * DM + 2 * lane + 1] = (_Float16)(d1 * rs * g1 + bb1);
    }
  }
  for (int i = threadIdx.x; i < ROWS * DM; i += 256)
    sT[i] = (_Float16)tokens[(size_t)e0 * SEQL * DM + i];
  __syncthreads();

  // 4 q-proj tiles + 40 kv-proj tiles
  for (int t = wave; t < 44; t += 8) {
    if (t < 4) {
      int nt = t;
      v8f acc = {};
#pragma unroll
      for (int kk = 0; kk < 64; kk += 32) {
        v16h a = load_a_frag(sQ, DM, 0, kk);
        v16h b = load_b_frag(Wci, DM, nt * 16, kk);
        acc = wmma16(a, b, acc);
      }
      int n = nt * 16 + (lane & 15);
      float bias = bci[n];
      int hh = lane >> 4;
      const float scale = 0.35355339059327373f;
#pragma unroll
      for (int r = 0; r < 8; ++r) sQp[(hh * 8 + r) * DM + n] = (acc[r] + bias) * scale;
    } else {
      int tt = t - 4;
      int mt = tt / 8, nt = tt % 8;
      v8f acc = {};
#pragma unroll
      for (int kk = 0; kk < 64; kk += 32) {
        v16h a = load_a_frag(sT, DM, mt * 16, kk);
        v16h b = load_b_frag(Wci, DM, 64 + nt * 16, kk);
        acc = wmma16(a, b, acc);
      }
      int n = nt * 16 + (lane & 15);
      float bias = bci[64 + n];
      int hh = lane >> 4;
#pragma unroll
      for (int r = 0; r < 8; ++r) sKV[(mt * 16 + hh * 8 + r) * 128 + n] = (_Float16)(acc[r] + bias);
    }
  }
  __syncthreads();

  // per-(edge,head) 1x5 attention
  if (threadIdx.x < EB * NH) {
    int el = threadIdx.x >> 3, hh = threadIdx.x & 7;
    int e = e0 + el;
    float qv[DH];
#pragma unroll
    for (int j = 0; j < DH; ++j) qv[j] = sQp[el * DM + hh * DH + j];
    float sc[SEQL];
    float mx = -1e30f;
#pragma unroll
    for (int s = 0; s < SEQL; ++s) {
      bool valid = idx[e * SEQL + s] < N;
      float dd = 0.f;
#pragma unroll
      for (int j = 0; j < DH; ++j)
        dd += qv[j] * (float)sKV[(el * SEQL + s) * 128 + hh * DH + j];
      sc[s] = valid ? dd : -1e9f;
      mx = fmaxf(mx, sc[s]);
    }
    float den = 0.f;
#pragma unroll
    for (int s = 0; s < SEQL; ++s) { sc[s] = __expf(sc[s] - mx); den += sc[s]; }
    float inv = 1.f / den;
#pragma unroll
    for (int j = 0; j < DH; ++j) {
      float o = 0.f;
#pragma unroll
      for (int s = 0; s < SEQL; ++s)
        o += sc[s] * (float)sKV[(el * SEQL + s) * 128 + 64 + hh * DH + j];
      sCO[el * DM + hh * DH + j] = (_Float16)(o * inv);
    }
  }
  __syncthreads();

  // out-proj + inst residual -> sI
  if (wave < 4) {
    int nt = wave;
    v8f acc = {};
#pragma unroll
    for (int kk = 0; kk < 64; kk += 32) {
      v16h a = load_a_frag(sCO, DM, 0, kk);
      v16h b = load_b_frag(Wco, DM, nt * 16, kk);
      acc = wmma16(a, b, acc);
    }
    int n = nt * 16 + (lane & 15);
    float bias = bco[n];
    int hh = lane >> 4;
#pragma unroll
    for (int r = 0; r < 8; ++r) {
      int m = hh * 8 + r;
      sI[m * DM + n] = inst[(size_t)(e0 + m) * DM + n] + acc[r] + bias;
    }
  }
  __syncthreads();
  // LN2 -> sA
  {
    const float g0 = ln2_g[2 * lane], g1 = ln2_g[2 * lane + 1];
    const float bb0 = ln2_b[2 * lane], bb1 = ln2_b[2 * lane + 1];
    for (int r = wave; r < EB; r += 8) {
      float x0 = sI[r * DM + 2 * lane], x1 = sI[r * DM + 2 * lane + 1];
      float m = wave_sum32(x0 + x1) * (1.f / 64.f);
      float d0 = x0 - m, d1 = x1 - m;
      float var = wave_sum32(d0 * d0 + d1 * d1) * (1.f / 64.f);
      float rs = rsqrtf(var + 1e-5f);
      sA[r * DM + 2 * lane]     = (_Float16)(d0 * rs * g0 + bb0);
      sA[r * DM + 2 * lane + 1] = (_Float16)(d1 * rs * g1 + bb1);
    }
  }
  __syncthreads();
  // FF1 (ca) with GELU
  for (int nt = wave; nt < 16; nt += 8) {
    v8f acc = {};
#pragma unroll
    for (int kk = 0; kk < 64; kk += 32) {
      v16h a = load_a_frag(sA, DM, 0, kk);
      v16h b = load_b_frag(W1, DM, nt * 16, kk);
      acc = wmma16(a, b, acc);
    }
    int n = nt * 16 + (lane & 15);
    float bias = b1[n];
    int hh = lane >> 4;
#pragma unroll
    for (int r = 0; r < 8; ++r) sH[(hh * 8 + r) * 256 + n] = (_Float16)gelu_erf(acc[r] + bias);
  }
  __syncthreads();
  // FF2 (ca) + residual -> inst (final)
  if (wave < 4) {
    int nt = wave;
    v8f acc = {};
#pragma unroll
    for (int kk = 0; kk < 256; kk += 32) {
      v16h a = load_a_frag(sH, 256, 0, kk);
      v16h b = load_b_frag(W2, 256, nt * 16, kk);
      acc = wmma16(a, b, acc);
    }
    int n = nt * 16 + (lane & 15);
    float bias = b2[n];
    int hh = lane >> 4;
#pragma unroll
    for (int r = 0; r < 8; ++r) {
      int m = hh * 8 + r;
      inst[(size_t)(e0 + m) * DM + n] = sI[m * DM + n] + acc[r] + bias;
    }
  }
}

// ---------------- head projection + leaky scores + segment max ----------------

__global__ __launch_bounds__(256) void eft_kernel(
    const float* __restrict__ inst, const _Float16* __restrict__ Wi,
    const float* __restrict__ attn_vec, const int* __restrict__ dst,
    float* __restrict__ a_out, unsigned* __restrict__ seg_max,
    _Float16* __restrict__ eft, int E) {
  __shared__ _Float16 sA[EB * DM];
  __shared__ _Float16 sE[EB * 512];
  const int e0 = blockIdx.x * EB;
  const int wave = threadIdx.x >> 5, lane = threadIdx.x & 31;
  for (int i = threadIdx.x; i < EB * DM; i += 256) sA[i] = (_Float16)inst[(size_t)e0 * DM + i];
  __syncthreads();
  for (int nt = wave; nt < 32; nt += 8) {
    v8f acc = {};
#pragma unroll
    for (int kk = 0; kk < 64; kk += 32) {
      v16h a = load_a_frag(sA, DM, 0, kk);
      v16h b = load_b_frag(Wi, DM, nt * 16, kk);
      acc = wmma16(a, b, acc);
    }
    int n = nt * 16 + (lane & 15);
    int hh = lane >> 4;
#pragma unroll
    for (int r = 0; r < 8; ++r) {
      int m = hh * 8 + r;
      _Float16 hv = (_Float16)acc[r];
      sE[m * 512 + n] = hv;
      eft[(size_t)(e0 + m) * 512 + n] = hv;
    }
  }
  __syncthreads();
  if (threadIdx.x < EB * NH) {
    int el = threadIdx.x >> 3, hh = threadIdx.x & 7;
    float s = 0.f;
#pragma unroll
    for (int c = 0; c < DM; ++c) s += (float)sE[el * 512 + hh * DM + c] * attn_vec[hh * DM + c];
    float av = leaky01(s);
    int e = e0 + el;
    a_out[(size_t)e * NH + hh] = av;
    atomicMax(&seg_max[(size_t)dst[e] * NH + hh], fenc(av));
  }
}

__global__ __launch_bounds__(256) void softmax_pass2_kernel(
    const float* __restrict__ a, const int* __restrict__ dst,
    const unsigned* __restrict__ seg_max, float* __restrict__ aexp,
    float* __restrict__ seg_sum, int E) {
  int i = blockIdx.x * 256 + threadIdx.x;
  if (i >= E * NH) return;
  int e = i >> 3, hh = i & 7;
  float m = fdec(seg_max[(size_t)dst[e] * NH + hh]);
  float ex = __expf(a[i] - m);
  aexp[i] = ex;
  atomicAdd(&seg_sum[(size_t)dst[e] * NH + hh], ex);
}

__global__ __launch_bounds__(256) void scatter_kernel(
    const _Float16* __restrict__ eft, const float* __restrict__ aexp,
    const float* __restrict__ seg_sum, const int* __restrict__ dst,
    float* __restrict__ out, int E) {
  long long i = (long long)blockIdx.x * 256 + threadIdx.x;
  if (i >= (long long)E * 512) return;
  int e = (int)(i >> 9);
  int r = (int)(i & 511);
  int hh = r >> 6;
  int d = dst[e];
  float att = aexp[(size_t)e * NH + hh] / seg_sum[(size_t)d * NH + hh];
  atomicAdd(&out[(size_t)d * 512 + r], (float)eft[i] * att);
}

// ---------------- host ----------------

extern "C" void kernel_launch(void* const* d_in, const int* in_sizes, int n_in,
                              void* d_out, int out_size, void* d_ws, size_t ws_size,
                              hipStream_t stream) {
  const float* features    = (const float*)d_in[0];
  const float* pos_emb     = (const float*)d_in[1];
  const float* self_in_w   = (const float*)d_in[2];
  const float* self_in_b   = (const float*)d_in[3];
  const float* self_out_w  = (const float*)d_in[4];
  const float* self_out_b  = (const float*)d_in[5];
  const float* cross_in_w  = (const float*)d_in[6];
  const float* cross_in_b  = (const float*)d_in[7];
  const float* cross_out_w = (const float*)d_in[8];
  const float* cross_out_b = (const float*)d_in[9];
  const float* sa_ln1_g = (const float*)d_in[10];
  const float* sa_ln1_b = (const float*)d_in[11];
  const float* sa_ln2_g = (const float*)d_in[12];
  const float* sa_ln2_b = (const float*)d_in[13];
  const float* ca_ln1_g = (const float*)d_in[14];
  const float* ca_ln1_b = (const float*)d_in[15];
  const float* ca_ln2_g = (const float*)d_in[16];
  const float* ca_ln2_b = (const float*)d_in[17];
  const float* sa_ff_w1 = (const float*)d_in[18];
  const float* sa_ff_b1 = (const float*)d_in[19];
  const float* sa_ff_w2 = (const float*)d_in[20];
  const float* sa_ff_b2 = (const float*)d_in[21];
  const float* ca_ff_w1 = (const float*)d_in[22];
  const float* ca_ff_b1 = (const float*)d_in[23];
  const float* ca_ff_w2 = (const float*)d_in[24];
  const float* ca_ff_b2 = (const float*)d_in[25];
  const float* inst_w   = (const float*)d_in[26];
  const float* attn_vec = (const float*)d_in[27];
  const int* idx = (const int*)d_in[28];
  const int* dst = (const int*)d_in[29];

  const int E = in_sizes[29];
  const int N = in_sizes[0] / DM;
  float* out = (float*)d_out;

  // workspace carve-up (256B aligned)
  size_t off = 0;
  auto take = [&](size_t bytes) -> char* {
    char* p = (char*)d_ws + off;
    off += (bytes + 255) & ~(size_t)255;
    return p;
  };
  _Float16* wh     = (_Float16*)take((size_t)131072 * sizeof(_Float16));
  float*    tokens = (float*)take((size_t)E * SEQL * DM * sizeof(float));
  float*    inst   = (float*)take((size_t)E * DM * sizeof(float));
  _Float16* eft    = (_Float16*)take((size_t)E * 512 * sizeof(_Float16));
  float*    ascore = (float*)take((size_t)E * NH * sizeof(float));
  float*    aexp   = (float*)take((size_t)E * NH * sizeof(float));
  unsigned* segmax = (unsigned*)take((size_t)N * NH * sizeof(unsigned));
  float*    segsum = (float*)take((size_t)N * NH * sizeof(float));

  // fp16 weight repack offsets (halves)
  _Float16* wh_self_in  = wh + 0;       // [192,64]
  _Float16* wh_self_out = wh + 12288;   // [64,64]
  _Float16* wh_cross_in = wh + 16384;   // [192,64]
  _Float16* wh_cross_out= wh + 28672;   // [64,64]
  _Float16* wh_sa_w1    = wh + 32768;   // [256,64]
  _Float16* wh_sa_w2    = wh + 49152;   // [64,256]
  _Float16* wh_ca_w1    = wh + 65536;   // [256,64]
  _Float16* wh_ca_w2    = wh + 81920;   // [64,256]
  _Float16* wh_inst     = wh + 98304;   // [512,64]

  cvt_kernel<<<(192 * 64 + 255) / 256, 256, 0, stream>>>(self_in_w, wh_self_in, 192 * 64);
  cvt_kernel<<<(64 * 64 + 255) / 256, 256, 0, stream>>>(self_out_w, wh_self_out, 64 * 64);
  cvt_kernel<<<(192 * 64 + 255) / 256, 256, 0, stream>>>(cross_in_w, wh_cross_in, 192 * 64);
  cvt_kernel<<<(64 * 64 + 255) / 256, 256, 0, stream>>>(cross_out_w, wh_cross_out, 64 * 64);
  cvt_kernel<<<(256 * 64 + 255) / 256, 256, 0, stream>>>(sa_ff_w1, wh_sa_w1, 256 * 64);
  cvt_kernel<<<(64 * 256 + 255) / 256, 256, 0, stream>>>(sa_ff_w2, wh_sa_w2, 64 * 256);
  cvt_kernel<<<(256 * 64 + 255) / 256, 256, 0, stream>>>(ca_ff_w1, wh_ca_w1, 256 * 64);
  cvt_kernel<<<(64 * 256 + 255) / 256, 256, 0, stream>>>(ca_ff_w2, wh_ca_w2, 64 * 256);
  cvt_kernel<<<(512 * 64 + 255) / 256, 256, 0, stream>>>(inst_w, wh_inst, 512 * 64);

  long long outN = (long long)N * 512;
  zero_kernel<<<(unsigned)((outN + 255) / 256), 256, 0, stream>>>((unsigned*)out, outN);
  long long segN = (long long)N * NH;
  zero_kernel<<<(unsigned)((segN + 255) / 256), 256, 0, stream>>>(segmax, segN);
  zero_kernel<<<(unsigned)((segN + 255) / 256), 256, 0, stream>>>((unsigned*)segsum, segN);

  int tokElems = E * SEQL * DM;
  gather_tokens_kernel<<<(tokElems + 255) / 256, 256, 0, stream>>>(features, pos_emb, idx, tokens, E, N);

  int nblk = E / EB;  // E = 150000 -> 9375 exact
  self_attn_kernel<<<nblk, 256, 0, stream>>>(tokens, idx, wh_self_in, self_in_b,
                                             wh_self_out, self_out_b, sa_ln1_g, sa_ln1_b, E, N);
  pool_ff_kernel<<<nblk, 256, 0, stream>>>(tokens, idx, inst, sa_ln2_g, sa_ln2_b,
                                           wh_sa_w1, sa_ff_b1, wh_sa_w2, sa_ff_b2, E, N);
  cross_ff_kernel<<<nblk, 256, 0, stream>>>(tokens, features, idx, inst,
                                            ca_ln1_g, ca_ln1_b, wh_cross_in, cross_in_b,
                                            wh_cross_out, cross_out_b, ca_ln2_g, ca_ln2_b,
                                            wh_ca_w1, ca_ff_b1, wh_ca_w2, ca_ff_b2, E, N);
  eft_kernel<<<nblk, 256, 0, stream>>>(inst, wh_inst, attn_vec, dst, ascore, segmax, eft, E);

  softmax_pass2_kernel<<<(E * NH + 255) / 256, 256, 0, stream>>>(ascore, dst, segmax, aexp, segsum, E);
  long long scatN = (long long)E * 512;
  scatter_kernel<<<(unsigned)((scatN + 255) / 256), 256, 0, stream>>>(eft, aexp, segsum, dst, out, E);
}
